// AttentionCircuit_28106265985564
// MI455X (gfx1250) — matrix operand order repaired and checked by reading, not compiled, not run on hardware
//
#include <hip/hip_runtime.h>
#include <hip/hip_bf16.h>

// ---------------------------------------------------------------------------
// Problem constants (B,S,D,R,N,H = 2,2048,1024,64,16,16; DH=64; M=B*S=4096)
// ---------------------------------------------------------------------------
#define M_TOT 4096
#define DIM   1024
#define RNK   64
#define NEXP  16
#define NHEAD 16
#define DHEAD 64
#define SEQ   2048

typedef __attribute__((ext_vector_type(16))) __bf16 bf16x16;
typedef __attribute__((ext_vector_type(8)))  float  f32x8;

// ---------------------------------------------------------------------------
// CDNA5 async global->LDS staging (GLOBAL_LOAD_ASYNC_TO_LDS_B128, ASYNCcnt).
// The clang builtin exists but its pointer params use builtin-only address
// spaces, so we issue the instruction via inline asm (bridge-doc recommended
// portable path). VDST = per-lane LDS byte address (low 32 bits of generic
// pointer == LDS offset per the flat-aperture rules), VADDR = 64-bit global.
// ---------------------------------------------------------------------------
#if defined(__gfx1250__)
#define ASYNC_LDS 1
#else
#define ASYNC_LDS 0
#endif

__device__ __forceinline__ void stage_b128(void* lds_dst, const void* gsrc) {
#if ASYNC_LDS
  asm volatile("global_load_async_to_lds_b128 %0, %1, off"
               :
               : "v"((unsigned int)(unsigned long long)lds_dst),
                 "v"((unsigned long long)gsrc)
               : "memory");
#else
  *(uint4*)lds_dst = *(const uint4*)gsrc;
#endif
}

__device__ __forceinline__ void wait_async_all() {
#if ASYNC_LDS
  asm volatile("s_wait_asynccnt 0x0" ::: "memory");
#endif
}

// ---------------------------------------------------------------------------
// WMMA helpers (CDNA5: V_WMMA_F32_16X16X32_BF16, wave32)
// C/D layout: lane l -> col = l&15, half = l>>4; elem v -> row = v + 8*half
// A layout  : lane l -> row = l&15; elem i -> k = (i<8?0:16) + 8*half + (i&7)
// B layout  : staged transposed in LDS (BT[n][k]); elem i -> k = (i<8?0:16) + 8*half + (i&7)
// ---------------------------------------------------------------------------
__device__ __forceinline__ f32x8 wmma_bf16(bf16x16 a, bf16x16 b, f32x8 c) {
  return __builtin_amdgcn_wmma_f32_16x16x32_bf16(
      /*neg_a=*/false, a, /*neg_b=*/false, b,
      /*c_mod=*/(short)0, c, /*reuse_a=*/false, /*reuse_b=*/false);
}

__device__ __forceinline__ bf16x16 ldsA(const __bf16* A, int lda, int row0, int k0) {
  const int l = threadIdx.x & 31, half = l >> 4, m = row0 + (l & 15);
  bf16x16 f;
#pragma unroll
  for (int i = 0; i < 16; ++i) {
    int k = k0 + ((i < 8) ? 0 : 16) + half * 8 + (i & 7);
    f[i] = A[m * lda + k];
  }
  return f;
}

__device__ __forceinline__ bf16x16 ldsBT(const __bf16* BT, int ldb, int col0, int k0) {
  const int l = threadIdx.x & 31, half = l >> 4, n = col0 + (l & 15);
  bf16x16 f;
#pragma unroll
  for (int i = 0; i < 16; ++i) f[i] = BT[n * ldb + k0 + ((i < 8) ? 0 : 16) + half * 8 + (i & 7)];
  return f;
}

// ---------------------------------------------------------------------------
// fp32 -> bf16 conversion
// ---------------------------------------------------------------------------
__global__ void cvt_f32_bf16(const float* __restrict__ s, __bf16* __restrict__ d, int n) {
  int i = blockIdx.x * blockDim.x + threadIdx.x;
  if (i < n) d[i] = (__bf16)s[i];
}

// ---------------------------------------------------------------------------
// Compress pass 1: T[n][m][r] = sum_d X[m][d] * F[n][d][r]   (bf16 WMMA)
// grid (M/64, NEXP), block 128 (4 waves; wave = 16 rows x 64 cols)
// ---------------------------------------------------------------------------
__global__ void __launch_bounds__(128)
expert_gemm_xF(const __bf16* __restrict__ X,   // [4096][1024]
               const __bf16* __restrict__ F,   // [16][1024][64]
               float* __restrict__ T)          // [16][4096][64]
{
  __shared__ __align__(16) __bf16 As[64 * 32];
  __shared__ __align__(16) __bf16 Bs[64 * 32];   // BT layout [r][k]
  const int m0 = blockIdx.x * 64;
  const int n  = blockIdx.y;
  const int t  = threadIdx.x, w = t >> 5, l = t & 31;
  const int half = l >> 4, col = l & 15;
  f32x8 acc[4] = {};

  for (int kc = 0; kc < DIM / 32; ++kc) {
    const int kb = kc * 32;
#pragma unroll
    for (int ss = 0; ss < 2; ++ss) {             // A: 64x32, async -> LDS
      int s = t + ss * 128;
      int row = s >> 2, c = (s & 3) * 8;
      const __bf16* g = &X[(size_t)(m0 + row) * DIM + kb + c];
      stage_b128(&As[row * 32 + c], g);
      __builtin_prefetch(g + 32, 0, 1);          // next K chunk
    }
    union { uint4 u; __bf16 b[8]; } tmp[2];
#pragma unroll
    for (int ss = 0; ss < 2; ++ss) {             // B: F[n][kb+kk][r] (reg)
      int s = t + ss * 128;
      int kk = s >> 3, r8 = (s & 7) * 8;
      tmp[ss].u = *(const uint4*)&F[((size_t)n * DIM + kb + kk) * RNK + r8];
    }
#pragma unroll
    for (int ss = 0; ss < 2; ++ss) {             // transpose -> Bs[r][kk]
      int s = t + ss * 128;
      int kk = s >> 3, r8 = (s & 7) * 8;
#pragma unroll
      for (int j = 0; j < 8; ++j) Bs[(r8 + j) * 32 + kk] = tmp[ss].b[j];
    }
    wait_async_all();
    __syncthreads();
    bf16x16 a = ldsA(As, 32, w * 16, 0);
#pragma unroll
    for (int j = 0; j < 4; ++j)
      acc[j] = wmma_bf16(a, ldsBT(Bs, 32, j * 16, 0), acc[j]);
    __syncthreads();
  }
#pragma unroll
  for (int j = 0; j < 4; ++j)
#pragma unroll
    for (int v = 0; v < 8; ++v) {
      int row = w * 16 + v + 8 * half;
      T[((size_t)n * M_TOT + m0 + row) * RNK + j * 16 + col] = acc[j][v];
    }
}

// Compress pass 2: h[m][r] = sum_n w[m][n] * T[n][m][r]  (deterministic reduce)
__global__ void reduce_experts(const float* __restrict__ T, const float* __restrict__ wgt,
                               float* __restrict__ h) {
  int i = blockIdx.x * blockDim.x + threadIdx.x;
  if (i >= M_TOT * RNK) return;
  int m = i >> 6;
  float s = 0.f;
#pragma unroll
  for (int n = 0; n < NEXP; ++n)
    s += wgt[m * NEXP + n] * T[(size_t)n * M_TOT * RNK + i];
  h[i] = s;
}

// ---------------------------------------------------------------------------
// Restore: O[m][d] = sum_n w[m][n] * (h[m][:] @ Rp[n])[d]  -> bf16, scaled
// grid (M/64, D/64), block 128; experts looped in-register (no atomics)
// ---------------------------------------------------------------------------
__global__ void __launch_bounds__(128)
restore_kernel(const float* __restrict__ h,    // [4096][64]
               const __bf16* __restrict__ Rp,  // [16][64][1024]
               const float* __restrict__ wgt,  // [4096][16]
               __bf16* __restrict__ O,         // [4096][1024]
               float scale)
{
  __shared__ __align__(16) float  hs[64 * 64];
  __shared__ __align__(16) float  wsh[64 * 16];
  __shared__ __align__(16) __bf16 Bs[64 * 64];  // BT layout [d][r]
  const int m0 = blockIdx.x * 64, d0 = blockIdx.y * 64;
  const int t = threadIdx.x, w = t >> 5, l = t & 31;
  const int half = l >> 4, col = l & 15;
  const int mrow = w * 16 + (l & 15);
  f32x8 acc[4] = {};

#pragma unroll
  for (int ss = 0; ss < 8; ++ss) {              // h tile (contiguous, async)
    int s = t + ss * 128;
    stage_b128(&hs[s * 4], &h[(size_t)m0 * RNK + s * 4]);
  }
#pragma unroll
  for (int ss = 0; ss < 2; ++ss) {              // weight tile (async)
    int s = t + ss * 128;
    stage_b128(&wsh[s * 4], &wgt[(size_t)m0 * NEXP + s * 4]);
  }

  for (int n = 0; n < NEXP; ++n) {
    union { uint4 u; __bf16 b[8]; } tmp[4];
#pragma unroll
    for (int ss = 0; ss < 4; ++ss) {            // Rp[n][r][d0+..] (reg)
      int s = t + ss * 128;
      int r = s >> 3, d8 = (s & 7) * 8;
      tmp[ss].u = *(const uint4*)&Rp[((size_t)n * RNK + r) * DIM + d0 + d8];
    }
#pragma unroll
    for (int ss = 0; ss < 4; ++ss) {            // transpose -> Bs[d][r]
      int s = t + ss * 128;
      int r = s >> 3, d8 = (s & 7) * 8;
#pragma unroll
      for (int j = 0; j < 8; ++j) Bs[(d8 + j) * RNK + r] = tmp[ss].b[j];
    }
    wait_async_all();
    __syncthreads();
    const float wv = wsh[mrow * NEXP + n];
#pragma unroll
    for (int ks = 0; ks < RNK; ks += 32) {
      bf16x16 a;                                // A[m][r] = bf16(h*w) on the fly
#pragma unroll
      for (int i = 0; i < 16; ++i) {
        int k = ks + ((i < 8) ? 0 : 16) + half * 8 + (i & 7);
        a[i] = (__bf16)(hs[mrow * RNK + k] * wv);
      }
#pragma unroll
      for (int j = 0; j < 4; ++j)
        acc[j] = wmma_bf16(a, ldsBT(Bs, RNK, j * 16, ks), acc[j]);
    }
    __syncthreads();
  }
#pragma unroll
  for (int j = 0; j < 4; ++j)
#pragma unroll
    for (int v = 0; v < 8; ++v) {
      int row = w * 16 + v + 8 * half;
      O[(size_t)(m0 + row) * DIM + d0 + j * 16 + col] = (__bf16)(acc[j][v] * scale);
    }
}

// ---------------------------------------------------------------------------
// Causal flash attention per (b,h): grid (S/64, B*H), block 128 (4 waves).
// Each wave owns 16 query rows: scores(WMMA) -> online softmax -> P@V(WMMA).
// Q pre-scaled by 1/sqrt(DH) in restore.
// ---------------------------------------------------------------------------
__global__ void __launch_bounds__(128)
attn_kernel(const __bf16* __restrict__ Q, const __bf16* __restrict__ K,
            const __bf16* __restrict__ V, __bf16* __restrict__ AO)
{
  __shared__ __align__(16) __bf16 Qs[64 * 64];
  __shared__ __align__(16) __bf16 Ks[64 * 64];  // [key][dh] == BT for scores
  __shared__ __align__(16) __bf16 Vt[64 * 64];  // [dh][key] == BT for P@V
  __shared__ __align__(16) __bf16 Ps[64 * 64];  // A layout [q][key]
  __shared__ __align__(16) float  Ss[64 * 64];
  __shared__ float mb[64], lb[64], ab[64];
  const int qt = blockIdx.x, q0 = qt * 64;
  const int bb = blockIdx.y >> 4, hd = blockIdx.y & 15;
  const size_t rowbase = (size_t)bb * SEQ;
  const int t = threadIdx.x, w = t >> 5, l = t & 31;
  const int half = l >> 4, col = l & 15;

#pragma unroll
  for (int ss = 0; ss < 4; ++ss) {              // Q tile (async)
    int s = t + ss * 128;
    int r = s >> 3, c = (s & 7) * 8;
    stage_b128(&Qs[r * 64 + c], &Q[(rowbase + q0 + r) * DIM + hd * DHEAD + c]);
  }
  if (t < 64) { mb[t] = -1e30f; lb[t] = 0.f; }
  f32x8 o[4] = {};

  for (int kt = 0; kt <= qt; ++kt) {
    const int k0 = kt * 64;
    __syncthreads();                            // guard LDS reuse across tiles
    union { uint4 u; __bf16 b[8]; } tmp[4];
#pragma unroll
    for (int ss = 0; ss < 4; ++ss) {            // K async; V into regs
      int s = t + ss * 128;
      int r = s >> 3, c = (s & 7) * 8;
      stage_b128(&Ks[r * 64 + c], &K[(rowbase + k0 + r) * DIM + hd * DHEAD + c]);
      tmp[ss].u = *(const uint4*)&V[(rowbase + k0 + r) * DIM + hd * DHEAD + c];
    }
#pragma unroll
    for (int ss = 0; ss < 4; ++ss) {            // transpose V -> Vt[dh][key]
      int s = t + ss * 128;
      int r = s >> 3, c = (s & 7) * 8;
#pragma unroll
      for (int j = 0; j < 8; ++j) Vt[(c + j) * 64 + r] = tmp[ss].b[j];
    }
    wait_async_all();
    __syncthreads();

    f32x8 sc[4] = {};                           // S = Q @ K^T
#pragma unroll
    for (int ks = 0; ks < DHEAD; ks += 32) {
      bf16x16 a = ldsA(Qs, 64, w * 16, ks);
#pragma unroll
      for (int j = 0; j < 4; ++j)
        sc[j] = wmma_bf16(a, ldsBT(Ks, 64, j * 16, ks), sc[j]);
    }
#pragma unroll
    for (int j = 0; j < 4; ++j)
#pragma unroll
      for (int v = 0; v < 8; ++v) {             // causal mask + spill to LDS
        int r16 = v + 8 * half;
        int grow = q0 + w * 16 + r16, gcol = k0 + j * 16 + col;
        Ss[(w * 16 + r16) * 64 + j * 16 + col] = (gcol <= grow) ? sc[j][v] : -1e30f;
      }
    __syncthreads();

    if (l < 16) {                               // online softmax: 1 lane / row
      int lr = w * 16 + l;
      float mold = mb[lr], rm = mold;
      for (int c = 0; c < 64; ++c) rm = fmaxf(rm, Ss[lr * 64 + c]);
      float alpha = __expf(mold - rm), sum = 0.f;
      for (int c = 0; c < 64; ++c) {
        float p = __expf(Ss[lr * 64 + c] - rm);
        Ps[lr * 64 + c] = (__bf16)p;
        sum += p;
      }
      lb[lr] = alpha * lb[lr] + sum;
      ab[lr] = alpha; mb[lr] = rm;
    }
    __syncthreads();
#pragma unroll
    for (int j = 0; j < 4; ++j)                 // rescale running O
#pragma unroll
      for (int v = 0; v < 8; ++v) o[j][v] *= ab[w * 16 + v + 8 * half];
#pragma unroll
    for (int ks = 0; ks < 64; ks += 32) {       // O += P @ V
      bf16x16 a = ldsA(Ps, 64, w * 16, ks);
#pragma unroll
      for (int j = 0; j < 4; ++j)
        o[j] = wmma_bf16(a, ldsBT(Vt, 64, j * 16, ks), o[j]);
    }
  }
  __syncthreads();
#pragma unroll
  for (int j = 0; j < 4; ++j)
#pragma unroll
    for (int v = 0; v < 8; ++v) {
      int r16 = v + 8 * half;
      float inv = 1.0f / lb[w * 16 + r16];
      AO[(rowbase + q0 + w * 16 + r16) * DIM + hd * DHEAD + j * 16 + col] =
          (__bf16)(o[j][v] * inv);
    }
}

// ---------------------------------------------------------------------------
// Output projection: out[m][d] = sum_k AO[m][k] * W_O[d][k]  (fp32 out)
// W_O row-major [d][k] is already the BT layout.
// ---------------------------------------------------------------------------
__global__ void __launch_bounds__(128)
proj_kernel(const __bf16* __restrict__ A, const __bf16* __restrict__ Wt,
            float* __restrict__ out)
{
  __shared__ __align__(16) __bf16 As[64 * 32];
  __shared__ __align__(16) __bf16 Bs[64 * 32];
  const int m0 = blockIdx.x * 64, d0 = blockIdx.y * 64;
  const int t = threadIdx.x, w = t >> 5, l = t & 31;
  const int half = l >> 4, col = l & 15;
  f32x8 acc[4] = {};
  for (int kc = 0; kc < DIM / 32; ++kc) {
    int kb = kc * 32;
#pragma unroll
    for (int ss = 0; ss < 2; ++ss) {            // both tiles plain: async
      int s = t + ss * 128;
      int row = s >> 2, c = (s & 3) * 8;
      const __bf16* ga = &A[(size_t)(m0 + row) * DIM + kb + c];
      const __bf16* gb = &Wt[(size_t)(d0 + row) * DIM + kb + c];
      stage_b128(&As[row * 32 + c], ga);
      stage_b128(&Bs[row * 32 + c], gb);
      __builtin_prefetch(ga + 32, 0, 1);
      __builtin_prefetch(gb + 32, 0, 1);
    }
    wait_async_all();
    __syncthreads();
    bf16x16 a = ldsA(As, 32, w * 16, 0);
#pragma unroll
    for (int j = 0; j < 4; ++j)
      acc[j] = wmma_bf16(a, ldsBT(Bs, 32, j * 16, 0), acc[j]);
    __syncthreads();
  }
#pragma unroll
  for (int j = 0; j < 4; ++j)
#pragma unroll
    for (int v = 0; v < 8; ++v) {
      int row = w * 16 + v + 8 * half;
      out[(size_t)(m0 + row) * DIM + d0 + j * 16 + col] = acc[j][v];
    }
}

// ---------------------------------------------------------------------------
// Host launch
// ---------------------------------------------------------------------------
extern "C" void kernel_launch(void* const* d_in, const int* in_sizes, int n_in,
                              void* d_out, int out_size, void* d_ws, size_t ws_size,
                              hipStream_t stream) {
  const float* x   = (const float*)d_in[0];
  const float* fqw = (const float*)d_in[1];
  const float* fkw = (const float*)d_in[2];
  const float* fvw = (const float*)d_in[3];
  const float* rqw = (const float*)d_in[4];
  const float* rkw = (const float*)d_in[5];
  const float* rvw = (const float*)d_in[6];
  const float* f_q = (const float*)d_in[7];
  const float* f_k = (const float*)d_in[8];
  const float* f_v = (const float*)d_in[9];
  const float* r_q = (const float*)d_in[10];
  const float* r_k = (const float*)d_in[11];
  const float* r_v = (const float*)d_in[12];
  const float* w_o = (const float*)d_in[13];

  char* ws = (char*)d_ws;
  size_t off = 0;
  auto take = [&](size_t bytes) -> void* {
    void* p = ws + off; off += (bytes + 255) & ~(size_t)255; return p;
  };
  const size_t XB   = (size_t)M_TOT * DIM * 2;        // 8 MB
  const size_t POOL = (size_t)NEXP * DIM * RNK * 2;   // 2 MB
  __bf16* xb  = (__bf16*)take(XB);
  __bf16* fqb = (__bf16*)take(POOL);
  __bf16* fkb = (__bf16*)take(POOL);
  __bf16* fvb = (__bf16*)take(POOL);
  __bf16* rqb = (__bf16*)take(POOL);
  __bf16* rkb = (__bf16*)take(POOL);
  __bf16* rvb = (__bf16*)take(POOL);
  __bf16* wob = (__bf16*)take((size_t)DIM * DIM * 2);
  float*  hq  = (float*)take((size_t)M_TOT * RNK * 4);
  float*  hk  = (float*)take((size_t)M_TOT * RNK * 4);
  float*  hv  = (float*)take((size_t)M_TOT * RNK * 4);
  __bf16* Qb  = (__bf16*)take(XB);
  __bf16* Kb  = (__bf16*)take(XB);
  __bf16* Vb  = (__bf16*)take(XB);
  __bf16* AO  = (__bf16*)take(XB);
  float*  T   = (float*)take((size_t)NEXP * M_TOT * RNK * 4);  // 16 MB scratch

  const int NX = M_TOT * DIM;            // 4194304
  const int NP = NEXP * DIM * RNK;       // 1048576 (== DIM*DIM)
  cvt_f32_bf16<<<(NX + 255) / 256, 256, 0, stream>>>(x,   xb,  NX);
  cvt_f32_bf16<<<(NP + 255) / 256, 256, 0, stream>>>(f_q, fqb, NP);
  cvt_f32_bf16<<<(NP + 255) / 256, 256, 0, stream>>>(f_k, fkb, NP);
  cvt_f32_bf16<<<(NP + 255) / 256, 256, 0, stream>>>(f_v, fvb, NP);
  cvt_f32_bf16<<<(NP + 255) / 256, 256, 0, stream>>>(r_q, rqb, NP);
  cvt_f32_bf16<<<(NP + 255) / 256, 256, 0, stream>>>(r_k, rkb, NP);
  cvt_f32_bf16<<<(NP + 255) / 256, 256, 0, stream>>>(r_v, rvb, NP);
  cvt_f32_bf16<<<(NP + 255) / 256, 256, 0, stream>>>(w_o, wob, NP);

  const dim3 gG(M_TOT / 64, NEXP);
  const int  NH = M_TOT * RNK;
  // compress q/k/v (reuse T scratch sequentially; stream serializes)
  expert_gemm_xF<<<gG, 128, 0, stream>>>(xb, fqb, T);
  reduce_experts<<<(NH + 255) / 256, 256, 0, stream>>>(T, fqw, hq);
  expert_gemm_xF<<<gG, 128, 0, stream>>>(xb, fkb, T);
  reduce_experts<<<(NH + 255) / 256, 256, 0, stream>>>(T, fkw, hk);
  expert_gemm_xF<<<gG, 128, 0, stream>>>(xb, fvb, T);
  reduce_experts<<<(NH + 255) / 256, 256, 0, stream>>>(T, fvw, hv);

  const dim3 gR(M_TOT / 64, DIM / 64);
  restore_kernel<<<gR, 128, 0, stream>>>(hq, rqb, rqw, Qb, 0.125f);  // 1/sqrt(64)
  restore_kernel<<<gR, 128, 0, stream>>>(hk, rkb, rkw, Kb, 1.0f);
  restore_kernel<<<gR, 128, 0, stream>>>(hv, rvb, rvw, Vb, 1.0f);

  attn_kernel<<<dim3(SEQ / 64, 2 * NHEAD), 128, 0, stream>>>(Qb, Kb, Vb, AO);

  proj_kernel<<<dim3(M_TOT / 64, DIM / 64), 128, 0, stream>>>(AO, wob, (float*)d_out);
}
